// SelectiveSSM_28939489640959
// MI455X (gfx1250) — matrix operand order repaired and checked by reading, not compiled, not run on hardware
//
#include <hip/hip_runtime.h>
#include <hip/hip_bf16.h>
#include <cstdint>
#include <cstddef>

typedef __bf16 bf16;
typedef __bf16 v16bf __attribute__((ext_vector_type(16)));
typedef float  v8f   __attribute__((ext_vector_type(8)));
typedef uint32_t u32x4 __attribute__((ext_vector_type(4)));
typedef uint32_t u32x8 __attribute__((ext_vector_type(8)));

#define DM   1024
#define NST  64
#define SEQL 2048
#define NB   2
#define NTOK (NB * SEQL)

union Frag32B { v16bf v; int4 q[2]; };

__device__ __forceinline__ float sig_(float x) { return 1.f / (1.f + __expf(-x)); }

// -------------------- f32 -> bf16 convert --------------------
__global__ void k_f32_to_bf16(const float* __restrict__ in, bf16* __restrict__ out, int n) {
  int i = blockIdx.x * blockDim.x + threadIdx.x;
  if (i < n) out[i] = (bf16)in[i];
}

// -------------------- LayerNorm (one block per token, 256 thr) --------------------
__global__ __launch_bounds__(256) void k_layernorm(const float* __restrict__ x,
                                                   const float* __restrict__ g,
                                                   const float* __restrict__ bta,
                                                   bf16* __restrict__ out) {
  int tok = blockIdx.x;
  int tid = threadIdx.x;
  const float4* xp = (const float4*)(x + (size_t)tok * DM);
  float4 v = xp[tid];
  float s  = v.x + v.y + v.z + v.w;
  float ss = v.x * v.x + v.y * v.y + v.z * v.z + v.w * v.w;
#pragma unroll
  for (int off = 16; off > 0; off >>= 1) {
    s  += __shfl_down(s, off, 32);
    ss += __shfl_down(ss, off, 32);
  }
  __shared__ float red[18];
  int lane = tid & 31, wv = tid >> 5;
  if (lane == 0) { red[wv] = s; red[8 + wv] = ss; }
  __syncthreads();
  if (tid == 0) {
    float ts = 0.f, tss = 0.f;
    for (int i = 0; i < 8; i++) { ts += red[i]; tss += red[8 + i]; }
    float mu = ts * (1.f / DM);
    red[16] = mu;
    red[17] = rsqrtf(tss * (1.f / DM) - mu * mu + 1e-5f);
  }
  __syncthreads();
  float mu = red[16], rstd = red[17];
  int d = tid * 4;
  bf16* op = out + (size_t)tok * DM + d;
  op[0] = (bf16)((v.x - mu) * rstd * g[d + 0] + bta[d + 0]);
  op[1] = (bf16)((v.y - mu) * rstd * g[d + 1] + bta[d + 1]);
  op[2] = (bf16)((v.z - mu) * rstd * g[d + 2] + bta[d + 2]);
  op[3] = (bf16)((v.w - mu) * rstd * g[d + 3] + bta[d + 3]);
}

// -------------------- TDM: DMA one 2D panel (tile_rows x tile_cols) into LDS --------------------
// D# per CDNA5 ISA. Groups 2/3 zeroed and passed twice (allowed per ISA).
__device__ __forceinline__ void tdm_load_panel(uint32_t lds_off, const void* gptr,
                                               uint32_t rows_total, uint32_t row_len,
                                               uint32_t tile_rows, uint32_t tile_cols) {
  uint64_t ga = (uint64_t)(uintptr_t)gptr;
  u32x4 g0;
  g0[0] = 1u;                                    // count = 1 (valid descriptor)
  g0[1] = lds_off;                               // lds_addr
  g0[2] = (uint32_t)ga;                          // global_addr[31:0]
  g0[3] = (uint32_t)(ga >> 32) | (2u << 30);     // global_addr[56:32] | type=2 (image)
  u32x8 g1;
  g1[0] = (1u << 16);                            // workgroup_mask=0, data_size=1 (2 bytes)
  g1[1] = (row_len & 0xffffu) << 16;             // tensor_dim0[15:0]  @bit48
  g1[2] = (row_len >> 16) | ((rows_total & 0xffffu) << 16);  // dim0[31:16] | dim1[15:0]
  g1[3] = (rows_total >> 16) | ((tile_cols & 0xffffu) << 16);// dim1[31:16] | tile_dim0
  g1[4] = tile_rows & 0xffffu;                   // tile_dim1 (tile_dim2 = 0)
  g1[5] = row_len;                               // tensor_dim0_stride[31:0]
  g1[6] = 0u;                                    // stride hi | dim1_stride lo
  g1[7] = 0u;
  u32x4 gz = {0u, 0u, 0u, 0u};
  asm volatile("tensor_load_to_lds %0, %1, %2, %3"
               :: "s"(g0), "s"(g1), "s"(gz), "s"(gz)
               : "memory");
}

// -------------------- TDM-staged bf16 WMMA GEMM: C[M,N] = A[M,K]*W[N,K]^T (+addend) --------
// Block 256 thr = 8 waves; block tile 64(M) x 64(N), full-K panels DMA'd to LDS by the TDM.
// LDS: 2 * 64 * K * 2B (K=1024 -> 256KB, dynamic). Inner loop = ds_load_b128 + v_wmma only.
__global__ __launch_bounds__(256) void k_gemm_bf16_tdm(const bf16* __restrict__ A,
                                                       const bf16* __restrict__ W,
                                                       float* __restrict__ C,
                                                       const float* __restrict__ addend,
                                                       int M, int N, int K) {
  extern __shared__ bf16 smem[];
  bf16* sA = smem;                 // [64][K]
  bf16* sW = smem + (size_t)64 * K;
  int m0b = blockIdx.y * 64;
  int n0b = blockIdx.x * 64;

  if ((threadIdx.x >> 5) == 0) {   // wave 0 issues both DMAs (TENSORcnt is per-wave)
    tdm_load_panel((uint32_t)(uintptr_t)sA, A + (size_t)m0b * K, (uint32_t)M, (uint32_t)K,
                   64u, (uint32_t)K);
    tdm_load_panel((uint32_t)(uintptr_t)sW, W + (size_t)n0b * K, (uint32_t)N, (uint32_t)K,
                   64u, (uint32_t)K);
    __builtin_amdgcn_s_wait_tensorcnt(0);
  }
  __syncthreads();

  int lane = threadIdx.x & 31;
  int wave = threadIdx.x >> 5;
  int wm = wave & 3;               // 4 waves down M
  int wn = wave >> 2;              // 2 waves across N
  int half = lane >> 4;
  int l16  = lane & 15;

  const bf16* Ap  = sA + (size_t)(wm * 16 + l16) * K + half * 16;
  const bf16* Wp0 = sW + (size_t)(wn * 32 + l16) * K + half * 16;
  const bf16* Wp1 = sW + (size_t)(wn * 32 + 16 + l16) * K + half * 16;

  v8f acc0 = {}, acc1 = {};
  for (int k = 0; k < K; k += 32) {
    Frag32B a, b0, b1;
    a.q[0]  = *(const int4*)(Ap + k);
    a.q[1]  = *(const int4*)(Ap + k + 8);
    b0.q[0] = *(const int4*)(Wp0 + k);
    b0.q[1] = *(const int4*)(Wp0 + k + 8);
    b1.q[0] = *(const int4*)(Wp1 + k);
    b1.q[1] = *(const int4*)(Wp1 + k + 8);
    acc0 = __builtin_amdgcn_wmma_f32_16x16x32_bf16(false, a.v, false, b0.v, (short)0, acc0, false, false);
    acc1 = __builtin_amdgcn_wmma_f32_16x16x32_bf16(false, a.v, false, b1.v, (short)0, acc1, false, false);
  }
  int mrow = m0b + wm * 16 + half * 8;
  int nc = n0b + wn * 32 + l16;
#pragma unroll
  for (int v = 0; v < 8; v++) {
    size_t i0 = (size_t)(mrow + v) * N + nc;
    float r0 = acc0[v], r1 = acc1[v];
    if (addend) { r0 += addend[i0]; r1 += addend[i0 + 16]; }
    C[i0] = r0;
    C[i0 + 16] = r1;
  }
}

// -------------------- direct-global bf16 WMMA GEMM (for small N=64 projections) ----------
__global__ __launch_bounds__(256) void k_gemm_bf16_nt(const bf16* __restrict__ A,
                                                      const bf16* __restrict__ W,
                                                      float* __restrict__ C,
                                                      const float* __restrict__ addend,
                                                      int M, int N, int K) {
  int lane = threadIdx.x & 31;
  int wave = threadIdx.x >> 5;
  int wm = wave & 3;
  int wn = wave >> 2;
  int m0 = blockIdx.y * 64 + wm * 16;
  int n0 = blockIdx.x * 64 + wn * 32;
  int half = lane >> 4;
  int l16  = lane & 15;

  const bf16* Ap  = A + (size_t)(m0 + l16) * K + half * 16;
  const bf16* Wp0 = W + (size_t)(n0 + l16) * K + half * 16;
  const bf16* Wp1 = W + (size_t)(n0 + 16 + l16) * K + half * 16;

  v8f acc0 = {}, acc1 = {};
  for (int k = 0; k < K; k += 32) {
    Frag32B a, b0, b1;
    a.q[0]  = *(const int4*)(Ap + k);
    a.q[1]  = *(const int4*)(Ap + k + 8);
    b0.q[0] = *(const int4*)(Wp0 + k);
    b0.q[1] = *(const int4*)(Wp0 + k + 8);
    b1.q[0] = *(const int4*)(Wp1 + k);
    b1.q[1] = *(const int4*)(Wp1 + k + 8);
    acc0 = __builtin_amdgcn_wmma_f32_16x16x32_bf16(false, a.v, false, b0.v, (short)0, acc0, false, false);
    acc1 = __builtin_amdgcn_wmma_f32_16x16x32_bf16(false, a.v, false, b1.v, (short)0, acc1, false, false);
  }
  int mrow = m0 + half * 8;
  int nc = n0 + l16;
#pragma unroll
  for (int v = 0; v < 8; v++) {
    size_t i0 = (size_t)(mrow + v) * N + nc;
    float r0 = acc0[v], r1 = acc1[v];
    if (addend) { r0 += addend[i0]; r1 += addend[i0 + 16]; }
    C[i0] = r0;
    C[i0 + 16] = r1;
  }
}

// -------------------- causal depthwise conv1d (K=4) + bias + SiLU --------------------
__global__ void k_conv_silu(const float* __restrict__ xz, const float* __restrict__ cw,
                            const float* __restrict__ cb, float* __restrict__ xcf,
                            bf16* __restrict__ xcb) {
  size_t i = (size_t)blockIdx.x * blockDim.x + threadIdx.x;  // over B*S*D
  int d  = (int)(i & (DM - 1));
  int bt = (int)(i >> 10);
  int t  = bt & (SEQL - 1);
  float acc = cb[d];
#pragma unroll
  for (int j = 0; j < 4; j++) {
    int tt = t - 3 + j;
    if (tt >= 0) acc += cw[d * 4 + j] * xz[(size_t)(bt - 3 + j) * (2 * DM) + d];
  }
  float sv = acc * sig_(acc);
  xcf[i] = sv;
  xcb[i] = (bf16)sv;
}

// -------------------- softplus(dt_pre + b_dt), in place --------------------
__global__ void k_softplus_bias(float* __restrict__ dt, const float* __restrict__ bdt) {
  size_t i = (size_t)blockIdx.x * blockDim.x + threadIdx.x;
  int d = (int)(i & (DM - 1));
  float v = dt[i] + bdt[d];
  dt[i] = (v > 20.f) ? v : log1pf(__expf(v));
}

// -------------------- selective scan, fused with SiLU(z) gating --------------------
__global__ __launch_bounds__(256) void k_scan(const float* __restrict__ dt,
                                              const float* __restrict__ xc,
                                              const float* __restrict__ Bm,
                                              const float* __restrict__ Cm,
                                              const float* __restrict__ xz,
                                              const float* __restrict__ A_log,
                                              const float* __restrict__ Dp,
                                              bf16* __restrict__ ygate) {
  int b    = blockIdx.x >> 7;
  int dblk = blockIdx.x & 127;
  int dbase = dblk * 8;
  int tid  = threadIdx.x;
  int dl   = tid >> 5;
  int lane = tid & 31;
  int d = dbase + dl;

  float A0 = -__expf(A_log[lane]);
  float A1 = -__expf(A_log[lane + 32]);
  float dpv = Dp[d];
  float h0 = 0.f, h1 = 0.f;

  __shared__ float sB[64][NST];
  __shared__ float sC[64][NST];
  __shared__ float sdt[8][64];
  __shared__ float sxc[8][64];
  __shared__ float szg[8][64];

  const size_t basBC = (size_t)b * SEQL * NST;

  for (int tc = 0; tc < SEQL; tc += 64) {
    __syncthreads();
    {
      const float4* bp = (const float4*)(Bm + basBC + (size_t)tc * NST);
      const float4* cp = (const float4*)(Cm + basBC + (size_t)tc * NST);
      float4* sb4 = (float4*)&sB[0][0];
      float4* sc4 = (float4*)&sC[0][0];
#pragma unroll
      for (int r = 0; r < 4; r++) {
        sb4[tid + r * 256] = bp[tid + r * 256];
        sc4[tid + r * 256] = cp[tid + r * 256];
      }
    }
#pragma unroll
    for (int r = 0; r < 2; r++) {
      int e = tid + r * 256;
      int t = e >> 3, dd = e & 7;
      size_t row = (size_t)(b * SEQL + tc + t);
      sdt[dd][t] = dt[row * DM + dbase + dd];
      sxc[dd][t] = xc[row * DM + dbase + dd];
      szg[dd][t] = xz[row * (2 * DM) + DM + dbase + dd];
    }
    __syncthreads();

    if (tc + 64 < SEQL) {
      __builtin_prefetch(Bm + basBC + (size_t)(tc + 64) * NST + tid * 16, 0, 0);
      __builtin_prefetch(Cm + basBC + (size_t)(tc + 64) * NST + tid * 16, 0, 0);
    }

    for (int t = 0; t < 64; t++) {
      float dtv = sdt[dl][t];
      float xcv = sxc[dl][t];
      h0 = __expf(A0 * dtv) * h0 + xcv * sB[t][lane];
      h1 = __expf(A1 * dtv) * h1 + xcv * sB[t][lane + 32];
      float y = h0 * sC[t][lane] + h1 * sC[t][lane + 32];
#pragma unroll
      for (int off = 16; off > 0; off >>= 1) y += __shfl_down(y, off, 32);
      if (lane == 0) {
        float zv = szg[dl][t];
        float gv = zv * sig_(zv);
        ygate[(size_t)(b * SEQL + tc + t) * DM + d] = (bf16)((y + xcv * dpv) * gv);
      }
    }
  }
}

// -------------------- host-side orchestration --------------------
extern "C" void kernel_launch(void* const* d_in, const int* in_sizes, int n_in,
                              void* d_out, int out_size, void* d_ws, size_t ws_size,
                              hipStream_t stream) {
  const float* x      = (const float*)d_in[0];
  const float* ln_g   = (const float*)d_in[1];
  const float* ln_b   = (const float*)d_in[2];
  const float* W_in   = (const float*)d_in[3];
  const float* conv_w = (const float*)d_in[4];
  const float* conv_b = (const float*)d_in[5];
  const float* A_log  = (const float*)d_in[6];
  const float* Dp     = (const float*)d_in[7];
  const float* W_dt   = (const float*)d_in[8];
  const float* b_dt   = (const float*)d_in[9];
  const float* W_B    = (const float*)d_in[10];
  const float* W_C    = (const float*)d_in[11];
  const float* W_out  = (const float*)d_in[12];

  char* p = (char*)d_ws;
  auto alloc = [&](size_t bytes) -> void* {
    void* r = (void*)p;
    p += (bytes + 255) & ~(size_t)255;
    return r;
  };
  bf16*  xn_bf    = (bf16*)alloc((size_t)NTOK * DM * 2);
  bf16*  Win_bf   = (bf16*)alloc((size_t)2 * DM * DM * 2);
  bf16*  Wdt_bf   = (bf16*)alloc((size_t)DM * DM * 2);
  bf16*  WB_bf    = (bf16*)alloc((size_t)NST * DM * 2);
  bf16*  WC_bf    = (bf16*)alloc((size_t)NST * DM * 2);
  bf16*  Wout_bf  = (bf16*)alloc((size_t)DM * DM * 2);
  float* xz       = (float*)alloc((size_t)NTOK * 2 * DM * 4);
  float* xconv    = (float*)alloc((size_t)NTOK * DM * 4);
  bf16*  xconv_bf = (bf16*)alloc((size_t)NTOK * DM * 2);
  float* dtbuf    = (float*)alloc((size_t)NTOK * DM * 4);
  float* Bmb      = (float*)alloc((size_t)NTOK * NST * 4);
  float* Cmb      = (float*)alloc((size_t)NTOK * NST * 4);
  bf16*  ygate    = (bf16*)alloc((size_t)NTOK * DM * 2);

  const size_t tdm_lds = (size_t)2 * 64 * DM * sizeof(bf16);  // 256 KB (<= 320 KB/WGP)
  (void)hipFuncSetAttribute((const void*)k_gemm_bf16_tdm,
                            hipFuncAttributeMaxDynamicSharedMemorySize, (int)tdm_lds);

  // weights -> bf16
  k_f32_to_bf16<<<(2 * DM * DM + 255) / 256, 256, 0, stream>>>(W_in, Win_bf, 2 * DM * DM);
  k_f32_to_bf16<<<(DM * DM + 255) / 256, 256, 0, stream>>>(W_dt, Wdt_bf, DM * DM);
  k_f32_to_bf16<<<(NST * DM + 255) / 256, 256, 0, stream>>>(W_B, WB_bf, NST * DM);
  k_f32_to_bf16<<<(NST * DM + 255) / 256, 256, 0, stream>>>(W_C, WC_bf, NST * DM);
  k_f32_to_bf16<<<(DM * DM + 255) / 256, 256, 0, stream>>>(W_out, Wout_bf, DM * DM);

  // LayerNorm -> bf16
  k_layernorm<<<NTOK, 256, 0, stream>>>(x, ln_g, ln_b, xn_bf);

  // in-proj: [4096,2048] = xn * W_in^T  (TDM-staged)
  k_gemm_bf16_tdm<<<dim3(2 * DM / 64, NTOK / 64), 256, tdm_lds, stream>>>(
      xn_bf, Win_bf, xz, nullptr, NTOK, 2 * DM, DM);

  // causal depthwise conv + SiLU
  k_conv_silu<<<(NTOK * DM) / 256, 256, 0, stream>>>(xz, conv_w, conv_b, xconv, xconv_bf);

  // dt-proj (TDM-staged), then softplus
  k_gemm_bf16_tdm<<<dim3(DM / 64, NTOK / 64), 256, tdm_lds, stream>>>(
      xconv_bf, Wdt_bf, dtbuf, nullptr, NTOK, DM, DM);
  k_softplus_bias<<<(NTOK * DM) / 256, 256, 0, stream>>>(dtbuf, b_dt);

  // B/C projections (small N: direct-global WMMA)
  k_gemm_bf16_nt<<<dim3(NST / 64, NTOK / 64), 256, 0, stream>>>(
      xconv_bf, WB_bf, Bmb, nullptr, NTOK, NST, DM);
  k_gemm_bf16_nt<<<dim3(NST / 64, NTOK / 64), 256, 0, stream>>>(
      xconv_bf, WC_bf, Cmb, nullptr, NTOK, NST, DM);

  // selective scan + gate
  k_scan<<<NB * (DM / 8), 256, 0, stream>>>(dtbuf, xconv, Bmb, Cmb, xz, A_log, Dp, ygate);

  // out-proj + residual (TDM-staged)
  k_gemm_bf16_tdm<<<dim3(DM / 64, NTOK / 64), 256, tdm_lds, stream>>>(
      ygate, Wout_bf, (float*)d_out, x, NTOK, DM, DM);
}
